// MultiHeadAttentionbyHand_14929306321628
// MI455X (gfx1250) — matrix operand order repaired and checked by reading, not compiled, not run on hardware
//
#include <hip/hip_runtime.h>
#include <hip/hip_bf16.h>

// ---------------------------------------------------------------------------
// Fused MHA block for MI455X (gfx1250, wave32).
//   * all GEMMs on v_wmma_f32_16x16x32_f16 (fp32 accum)
//   * flash-style attention: scores never hit HBM
//   * CDNA5 async path: global_load_async_to_lds_b128 + s_wait_asynccnt,
//     double-buffered LDS tiles shared by all 4 waves of a block
// B=4, S=2048, H=1024, NH=16, HD=64  (~137 GFLOP, compute-bound)
// ---------------------------------------------------------------------------

typedef __attribute__((ext_vector_type(16))) _Float16 v16h;
typedef __attribute__((ext_vector_type(8)))  _Float16 v8h;
typedef __attribute__((ext_vector_type(8)))  float    v8f;

static constexpr int Bn  = 4;
static constexpr int Sn  = 2048;
static constexpr int Hn  = 1024;
static constexpr int NHn = 16;
static constexpr int HDn = 64;
static constexpr int BSn = Bn * Sn;      // 8192

#define WMMA16(a, b, c) \
  __builtin_amdgcn_wmma_f32_16x16x32_f16(false, (a), false, (b), (short)0, (c), false, false)

__device__ __forceinline__ v8f zero8() {
  v8f z = {0.f, 0.f, 0.f, 0.f, 0.f, 0.f, 0.f, 0.f};
  return z;
}

// A fragment (16x32 f16): lane = M row; lanes 0-15 K{0-7,16-23}, 16-31 K{8-15,24-31}
__device__ __forceinline__ v16h ldA(const _Float16* p, int ld) {
  int lane = threadIdx.x & 31;
  const _Float16* q = p + (size_t)(lane & 15) * ld + ((lane >> 4) << 3);
  union { v16h v; v8h h[2]; } u;
  u.h[0] = *(const v8h*)(q);
  u.h[1] = *(const v8h*)(q + 16);
  return u.v;
}

// B fragment (32x16 f16): lane = N column; memory must be N x K row-major.
// Lanes 0-15 hold K{0-7,16-23}, lanes 16-31 hold K{8-15,24-31}.
__device__ __forceinline__ v16h ldB(const _Float16* p, int ld) {
  int lane = threadIdx.x & 31;
  const _Float16* q = p + (size_t)(lane & 15) * ld + ((lane >> 4) << 3);
  union { v16h v; v8h h[2]; } u;
  u.h[0] = *(const v8h*)(q);
  u.h[1] = *(const v8h*)(q + 16);
  return u.v;
}

// ---- CDNA5 async copy: 16B global -> LDS, tracked by ASYNCcnt -------------
__device__ __forceinline__ unsigned lds_off(const void* p) {
  return (unsigned)(unsigned long long)p;   // low dword of LDS aperture addr
}
__device__ __forceinline__ void async_b128(unsigned lds, const void* g) {
  asm volatile("global_load_async_to_lds_b128 %0, %1, off"
               :: "v"(lds), "v"(g) : "memory");
}
__device__ __forceinline__ void wait_async0() {
  asm volatile("s_wait_asynccnt 0" ::: "memory");
}

// --------------------------- prep kernels ----------------------------------

__global__ __launch_bounds__(256) void cvt_f16_kernel(
    const float* __restrict__ in, _Float16* __restrict__ out, int n) {
  int i = blockIdx.x * 256 + threadIdx.x;
  if (i < n) out[i] = (_Float16)in[i];
}

// wt[z][n][k] = (f16) W_z[k][n]   (N x K layout for B-fragments)
__global__ __launch_bounds__(256) void transpose_w_kernel(
    const float* __restrict__ wq, const float* __restrict__ wk,
    const float* __restrict__ wv, const float* __restrict__ wd,
    _Float16* __restrict__ wt) {
  int z = blockIdx.y;
  const float* w = (z == 0) ? wq : (z == 1) ? wk : (z == 2) ? wv : wd;
  _Float16* o = wt + (size_t)z * Hn * Hn;
  int idx = blockIdx.x * 256 + threadIdx.x;
  int n = idx >> 10, k = idx & (Hn - 1);
  o[idx] = (_Float16)w[(size_t)k * Hn + n];
}

// --------------------------- shared GEMM mainloop --------------------------
// Block tile M=32, N=256, K-step 32; 4 waves, wave tile 32x64 (8 WMMA/step).
// Double-buffered LDS tiles filled by global_load_async_to_lds_b128.
__device__ __forceinline__ void gemm_mainloop(
    const _Float16* __restrict__ x,   // [BS,H] activations (row-major)
    const _Float16* __restrict__ w,   // [H,H]  weight^T    (N x K row-major)
    int m0, int n0blk, int wave,
    _Float16* As,                     // [2][32][32]
    _Float16* Bs,                     // [2][256][32]
    v8f acc[2][4])
{
  int t = threadIdx.x;
  int arow = t >> 2, aco = (t & 3) << 3;   // A: 128 x 16B chunks, 1/thread
  auto issue = [&](int kk, int buf) {
    async_b128(lds_off(As + buf * 1024 + arow * 32 + aco),
               x + (size_t)(m0 + arow) * Hn + kk + aco);
#pragma unroll
    for (int i = 0; i < 8; ++i) {          // B: 1024 x 16B chunks, 8/thread
      int c = t + i * 128;
      int row = c >> 2, co = (c & 3) << 3;
      async_b128(lds_off(Bs + buf * 8192 + row * 32 + co),
                 w + (size_t)(n0blk + row) * Hn + kk + co);
    }
  };

  issue(0, 0);
  int buf = 0;
  for (int kk = 0; kk < Hn; kk += 32) {
    wait_async0();        // our fills for this buffer are done
    __syncthreads();      // everyone's fills done / previous reads retired
    if (kk + 32 < Hn) issue(kk + 32, buf ^ 1);   // overlap next fill

    const _Float16* ap = As + buf * 1024;
    v16h af0 = ldA(ap, 32);
    v16h af1 = ldA(ap + 16 * 32, 32);
    const _Float16* bp0 = Bs + buf * 8192 + (wave * 64) * 32;
#pragma unroll
    for (int t4 = 0; t4 < 4; ++t4) {
      v16h bf = ldB(bp0 + t4 * 16 * 32, 32);
      acc[0][t4] = WMMA16(af0, bf, acc[0][t4]);
      acc[1][t4] = WMMA16(af1, bf, acc[1][t4]);
    }
    buf ^= 1;
  }
}

// --------------------------- QKV projection --------------------------------
// grid (BS/32, H/256, 3), block 128.
__global__ __launch_bounds__(128) void gemm_qkv_kernel(
    const _Float16* __restrict__ x,      // [BS,H] f16 query
    const _Float16* __restrict__ wts,    // [3][H][H] (WqT,WkT,WvT)
    const float* __restrict__ bq, const float* __restrict__ bk,
    const float* __restrict__ bv,
    _Float16* __restrict__ qo,           // [BS,H]
    _Float16* __restrict__ kqo,          // [BS,H]
    _Float16* __restrict__ vto)          // [B,NH,HD,S]  (transposed V)
{
  __shared__ __align__(16) _Float16 As[2 * 32 * 32];
  __shared__ __align__(16) _Float16 Bs[2 * 256 * 32];
  int which = blockIdx.z;
  const _Float16* w = wts + (size_t)which * Hn * Hn;
  const float* bias = (which == 0) ? bq : (which == 1) ? bk : bv;
  int wave = threadIdx.x >> 5, lane = threadIdx.x & 31;
  int m0 = blockIdx.x * 32;
  int n0 = blockIdx.y * 256 + wave * 64;

  v8f acc[2][4];
#pragma unroll
  for (int mi = 0; mi < 2; ++mi)
#pragma unroll
    for (int t4 = 0; t4 < 4; ++t4) acc[mi][t4] = zero8();

  gemm_mainloop(x, w, m0, blockIdx.y * 256, wave, As, Bs, acc);

  int mo = (lane >> 4) << 3, nl = lane & 15;
#pragma unroll
  for (int mi = 0; mi < 2; ++mi)
#pragma unroll
    for (int t4 = 0; t4 < 4; ++t4) {
      int col = n0 + t4 * 16 + nl;
      float bbias = bias[col];
      if (which < 2) {
        _Float16* o = (which == 0) ? qo : kqo;
#pragma unroll
        for (int r = 0; r < 8; ++r)
          o[(size_t)(m0 + mi * 16 + mo + r) * Hn + col] =
              (_Float16)(acc[mi][t4][r] + bbias);
      } else {
        int head = col >> 6, hd = col & 63;
#pragma unroll
        for (int r = 0; r < 8; ++r) {
          int m = m0 + mi * 16 + mo + r;
          int bidx = m >> 11, s = m & (Sn - 1);
          vto[((size_t)((bidx * NHn + head) * HDn + hd) << 11) + s] =
              (_Float16)(acc[mi][t4][r] + bbias);
        }
      }
    }
}

// --------------------------- flash attention -------------------------------
// grid (S/64, NH, B), block 128 (4 waves, one 16-row q tile each).
// K/V chunks (64x64) are shared by all 4 waves -> async-staged in LDS.
__global__ __launch_bounds__(128) void attn_kernel(
    const _Float16* __restrict__ q,   // [BS,H]
    const _Float16* __restrict__ k,   // [BS,H]
    const _Float16* __restrict__ vt,  // [B,NH,HD,S]
    const float* __restrict__ mask,   // [B,S]
    _Float16* __restrict__ ctx)       // [BS,H]
{
  __shared__ __align__(16) _Float16 Ks[2][64][64];   // rows = kv pos, cols = hd
  __shared__ __align__(16) _Float16 Vs[2][64][64];   // rows = hd, cols = kv pos
  __shared__ __align__(16) _Float16 plds[4][16][72];

  int wave = threadIdx.x >> 5, lane = threadIdx.x & 31;
  int b = blockIdx.z, head = blockIdx.y;
  int q0 = (blockIdx.x * 4 + wave) * 16;

  const _Float16* qp = q + (size_t)(b * Sn + q0) * Hn + head * HDn;
  const _Float16* kbase = k + (size_t)b * Sn * Hn + head * HDn;
  const _Float16* vp = vt + (size_t)(b * NHn + head) * HDn * Sn;
  const float* mp = mask + (size_t)b * Sn;

  v16h qa0 = ldA(qp, Hn);
  v16h qa1 = ldA(qp + 32, Hn);

  float rmax[8], rsum[8];
  v8f cacc[4];
#pragma unroll
  for (int r = 0; r < 8; ++r) { rmax[r] = -3.0e38f; rsum[r] = 0.f; }
#pragma unroll
  for (int t = 0; t < 4; ++t) cacc[t] = zero8();

  auto issue = [&](int kc2, int nbuf) {
#pragma unroll
    for (int i = 0; i < 4; ++i) {          // 512 x 16B chunks per tile
      int c = threadIdx.x + i * 128;
      int row = c >> 3, co = (c & 7) << 3;
      async_b128(lds_off(&Ks[nbuf][row][co]),
                 kbase + (size_t)(kc2 + row) * Hn + co);
      async_b128(lds_off(&Vs[nbuf][row][co]),
                 vp + (size_t)row * Sn + kc2 + co);
    }
  };

  issue(0, 0);
  int buf = 0;
  for (int kc = 0; kc < Sn; kc += 64) {
    wait_async0();
    __syncthreads();
    if (kc + 64 < Sn) issue(kc + 64, buf ^ 1);

    // ---- scores: 16 x 64, K = HD = 64
    v8f sc[4];
#pragma unroll
    for (int t = 0; t < 4; ++t) sc[t] = zero8();
#pragma unroll
    for (int t = 0; t < 4; ++t) {
      v16h b0 = ldB(&Ks[buf][t * 16][0], 64);
      v16h b1 = ldB(&Ks[buf][t * 16][32], 64);
      sc[t] = WMMA16(qa0, b0, sc[t]);
      sc[t] = WMMA16(qa1, b1, sc[t]);
    }
    // ---- scale + mask + online softmax
    float p[4][8], cmax[8];
#pragma unroll
    for (int r = 0; r < 8; ++r) cmax[r] = -3.0e38f;
#pragma unroll
    for (int t = 0; t < 4; ++t) {
      float mv = mp[kc + t * 16 + (lane & 15)];
#pragma unroll
      for (int r = 0; r < 8; ++r) {
        float sv = sc[t][r] * 0.125f + mv;   // 1/sqrt(64)
        p[t][r] = sv;
        cmax[r] = fmaxf(cmax[r], sv);
      }
    }
#pragma unroll
    for (int r = 0; r < 8; ++r)
#pragma unroll
      for (int m = 1; m < 16; m <<= 1)
        cmax[r] = fmaxf(cmax[r], __shfl_xor(cmax[r], m, 32));
    float corr[8], lsum[8];
#pragma unroll
    for (int r = 0; r < 8; ++r) {
      float nm = fmaxf(rmax[r], cmax[r]);
      corr[r] = __expf(rmax[r] - nm);
      rmax[r] = nm;
      lsum[r] = 0.f;
    }
#pragma unroll
    for (int t = 0; t < 4; ++t)
#pragma unroll
      for (int r = 0; r < 8; ++r) {
        float e = __expf(p[t][r] - rmax[r]);
        p[t][r] = e;
        lsum[r] += e;
      }
#pragma unroll
    for (int r = 0; r < 8; ++r) {
#pragma unroll
      for (int m = 1; m < 16; m <<= 1) lsum[r] += __shfl_xor(lsum[r], m, 32);
      rsum[r] = rsum[r] * corr[r] + lsum[r];
    }
#pragma unroll
    for (int t = 0; t < 4; ++t)
#pragma unroll
      for (int r = 0; r < 8; ++r) cacc[t][r] *= corr[r];

    // ---- P: C-fragment -> A-fragment via per-wave LDS tile
    int mrow = (lane >> 4) << 3, ncl = lane & 15;
#pragma unroll
    for (int t = 0; t < 4; ++t)
#pragma unroll
      for (int r = 0; r < 8; ++r)
        plds[wave][mrow + r][t * 16 + ncl] = (_Float16)p[t][r];
    asm volatile("s_wait_dscnt 0" ::: "memory");

    const _Float16* pb = &plds[wave][0][0];
    v16h pa0 = ldA(pb, 72);
    v16h pa1 = ldA(pb + 32, 72);

    // ---- ctx += P (16x64) x V (64x64) from staged V^T
#pragma unroll
    for (int t = 0; t < 4; ++t) {
      v16h v0 = ldB(&Vs[buf][t * 16][0], 64);
      v16h v1 = ldB(&Vs[buf][t * 16][32], 64);
      cacc[t] = WMMA16(pa0, v0, cacc[t]);
      cacc[t] = WMMA16(pa1, v1, cacc[t]);
    }
    buf ^= 1;
  }

  int mo = (lane >> 4) << 3, nl = lane & 15;
#pragma unroll
  for (int t = 0; t < 4; ++t) {
    int col = head * HDn + t * 16 + nl;
#pragma unroll
    for (int r = 0; r < 8; ++r)
      ctx[(size_t)(b * Sn + q0 + mo + r) * Hn + col] =
          (_Float16)(cacc[t][r] / rsum[r]);
  }
}

// --------------------------- output projection -----------------------------
// grid (BS/32, H/256), block 128.
__global__ __launch_bounds__(128) void gemm_out_kernel(
    const _Float16* __restrict__ x,   // ctx [BS,H]
    const _Float16* __restrict__ w,   // WdT [H][H]
    const float* __restrict__ bias, float* __restrict__ out)  // hidden f32
{
  __shared__ __align__(16) _Float16 As[2 * 32 * 32];
  __shared__ __align__(16) _Float16 Bs[2 * 256 * 32];
  int wave = threadIdx.x >> 5, lane = threadIdx.x & 31;
  int m0 = blockIdx.x * 32;
  int n0 = blockIdx.y * 256 + wave * 64;

  v8f acc[2][4];
#pragma unroll
  for (int mi = 0; mi < 2; ++mi)
#pragma unroll
    for (int t4 = 0; t4 < 4; ++t4) acc[mi][t4] = zero8();

  gemm_mainloop(x, w, m0, blockIdx.y * 256, wave, As, Bs, acc);

  int mo = (lane >> 4) << 3, nl = lane & 15;
#pragma unroll
  for (int mi = 0; mi < 2; ++mi)
#pragma unroll
    for (int t4 = 0; t4 < 4; ++t4) {
      int col = n0 + t4 * 16 + nl;
      float bbias = bias[col];
#pragma unroll
      for (int r = 0; r < 8; ++r)
        out[(size_t)(m0 + mi * 16 + mo + r) * Hn + col] = acc[mi][t4][r] + bbias;
    }
}

// --------------------------- LayerNorm + residual --------------------------
__global__ __launch_bounds__(256) void ln_res_kernel(
    const float* __restrict__ hidden, const float* __restrict__ query,
    const float* __restrict__ gamma, const float* __restrict__ beta,
    float* __restrict__ out)
{
  int row = blockIdx.x;
  const float* h = hidden + (size_t)row * Hn;
  float v[4], s = 0.f, ss = 0.f;
#pragma unroll
  for (int i = 0; i < 4; ++i) {
    v[i] = h[threadIdx.x + i * 256];
    s += v[i];
    ss += v[i] * v[i];
  }
#pragma unroll
  for (int m = 1; m < 32; m <<= 1) {
    s += __shfl_xor(s, m, 32);
    ss += __shfl_xor(ss, m, 32);
  }
  __shared__ float rs[8], rss[8];
  int w = threadIdx.x >> 5, ln = threadIdx.x & 31;
  if (ln == 0) { rs[w] = s; rss[w] = ss; }
  __syncthreads();
  if (w == 0) {
    float a = (ln < 8) ? rs[ln] : 0.f;
    float b = (ln < 8) ? rss[ln] : 0.f;
#pragma unroll
    for (int m = 1; m < 8; m <<= 1) {
      a += __shfl_xor(a, m, 32);
      b += __shfl_xor(b, m, 32);
    }
    if (ln == 0) { rs[0] = a; rss[0] = b; }
  }
  __syncthreads();
  float mu = rs[0] * (1.f / Hn);
  float var = fmaxf(rss[0] * (1.f / Hn) - mu * mu, 0.f);
  float inv = rsqrtf(var + 1e-12f);
  const float* qr = query + (size_t)row * Hn;
#pragma unroll
  for (int i = 0; i < 4; ++i) {
    int c = threadIdx.x + i * 256;
    out[(size_t)row * Hn + c] = (v[i] - mu) * inv * gamma[c] + beta[c] + qr[c];
  }
}

// --------------------------------------------------------------------------

extern "C" void kernel_launch(void* const* d_in, const int* in_sizes, int n_in,
                              void* d_out, int out_size, void* d_ws, size_t ws_size,
                              hipStream_t stream) {
  (void)in_sizes; (void)n_in; (void)out_size; (void)ws_size;
  const float* query = (const float*)d_in[0];
  const float* mask  = (const float*)d_in[1];
  const float* Wq = (const float*)d_in[2];
  const float* bq = (const float*)d_in[3];
  const float* Wk = (const float*)d_in[4];
  const float* bk = (const float*)d_in[5];
  const float* Wv = (const float*)d_in[6];
  const float* bv = (const float*)d_in[7];
  const float* Wd = (const float*)d_in[8];
  const float* bd = (const float*)d_in[9];
  const float* gamma = (const float*)d_in[10];
  const float* beta  = (const float*)d_in[11];
  float* out = (float*)d_out;

  // workspace layout (f16 elements); total 75,497,472 bytes
  _Float16* ws16 = (_Float16*)d_ws;
  _Float16* xf16 = ws16;                       //  8,388,608  (query f16; reused as ctx)
  _Float16* wt   = ws16 + 8388608;             //  4 x 1,048,576 (WqT WkT WvT WdT)
  _Float16* qb   = ws16 + 12582912;            //  8,388,608
  _Float16* kb   = ws16 + 20971520;            //  8,388,608
  _Float16* vtb  = ws16 + 29360128;            //  8,388,608
  _Float16* ctx  = xf16;                       //  alias (query f16 dead after QKV)
  float* hidden  = (float*)(ws16 + 20971520);  //  alias over k/vT (dead after attn)

  cvt_f16_kernel<<<BSn * Hn / 256, 256, 0, stream>>>(query, xf16, BSn * Hn);
  transpose_w_kernel<<<dim3(Hn * Hn / 256, 4), 256, 0, stream>>>(Wq, Wk, Wv, Wd, wt);
  gemm_qkv_kernel<<<dim3(BSn / 32, Hn / 256, 3), 128, 0, stream>>>(
      xf16, wt, bq, bk, bv, qb, kb, vtb);
  attn_kernel<<<dim3(Sn / 64, NHn, Bn), 128, 0, stream>>>(qb, kb, vtb, mask, ctx);
  gemm_out_kernel<<<dim3(BSn / 32, Hn / 256), 128, 0, stream>>>(
      ctx, wt + 3 * 1048576, bd, hidden);
  ln_res_kernel<<<BSn, 256, 0, stream>>>(hidden, query, gamma, beta, out);
}
